// DecoderPreNet_13692355739899
// MI455X (gfx1250) — matrix operand order repaired and checked by reading, not compiled, not run on hardware
//
#include <hip/hip_runtime.h>
#include <hip/hip_bf16.h>

typedef _Float16 v16h __attribute__((ext_vector_type(16)));
typedef _Float16 v8h  __attribute__((ext_vector_type(8)));
typedef float    v8f  __attribute__((ext_vector_type(8)));

#define B_ 8
#define H_ 8
#define LQ 2048
#define LK 1024
#define D_ 512
#define DK 64

union AFrag { v16h v; v8h h[2]; };

__device__ __forceinline__ v8f wmma_f16(v16h a, v16h b, v8f c) {
  // v_wmma_f32_16x16x32_f16, f32 accumulate
  return __builtin_amdgcn_wmma_f32_16x16x32_f16(false, a, false, b, (short)0, c, false, false);
}

// ---------------- fp32 -> fp16 convert ----------------
__global__ void k_cvt_f32_f16(const float* __restrict__ src, _Float16* __restrict__ dst, int n) {
  int i = blockIdx.x * blockDim.x + threadIdx.x;
  if (i < n) dst[i] = (_Float16)src[i];
}

// ---------------- gather pos_table rows + convert ----------------
__global__ void k_gather_pos(const float* __restrict__ tab, const int* __restrict__ pos,
                             _Float16* __restrict__ dst) {
  int i = blockIdx.x * blockDim.x + threadIdx.x;   // over B_*LQ*D_ (exact multiple of 256)
  int row = i >> 9;                                 // D_ == 512
  int d = i & 511;
  dst[i] = (_Float16)tab[(size_t)pos[row] * D_ + d];
}

// ---------------- generic projection: C = A @ W^T + bias ----------------
// A: [8*len, 512] f16 row-major.  W: [512, 512] f16 row-major (contraction over d).
// Output scattered to per-head layout via (headStride,rowStride,colStride).
// len == 1 << lenShift (power of two) -> no integer division in the store path.
__global__ void __launch_bounds__(128)
k_proj16(const _Float16* __restrict__ A, const _Float16* __restrict__ W,
         const float* __restrict__ bias, _Float16* __restrict__ out,
         int lenShift, int headStride, int rowStride, int colStride) {
  const int lane = threadIdx.x & 31, wib = threadIdx.x >> 5;
  const int w = blockIdx.x * 4 + wib;
  const int mtile = w >> 3, ng = w & 7;      // 8 column-groups of 64
  const int lane16 = lane & 15, grp = lane >> 4;

  v8f z = {};
  v8f o[4];
#pragma unroll
  for (int t = 0; t < 4; ++t) o[t] = z;

  const _Float16* ap0 = A + (size_t)(mtile * 16 + lane16) * D_;
  for (int c = 0; c < D_ / 32; ++c) {
    AFrag a;
    const _Float16* ap = ap0 + c * 32;
    a.h[0] = *(const v8h*)(ap + 16 * grp);
    a.h[1] = *(const v8h*)(ap + 16 * grp + 8);
#pragma unroll
    for (int t = 0; t < 4; ++t) {
      const _Float16* wp = W + (size_t)(ng * 64 + t * 16 + lane16) * D_ + c * 32 + 16 * grp;
      v16h bf = *(const v16h*)wp;
      o[t] = wmma_f16(a.v, bf, o[t]);
    }
  }

  const int lenMask = (1 << lenShift) - 1;
#pragma unroll
  for (int t = 0; t < 4; ++t) {
    int col = ng * 64 + t * 16 + lane16;
    int hh = col >> 6, dd = col & 63;
    float bv = bias[col];
#pragma unroll
    for (int i = 0; i < 8; ++i) {
      int m = mtile * 16 + i + 8 * grp;
      int bb = m >> lenShift, r = m & lenMask;
      out[(size_t)(bb * H_ + hh) * headStride + (size_t)r * rowStride + (size_t)dd * colStride] =
          (_Float16)(o[t][i] + bv);
    }
  }
}

// ---------------- flash attention: one wave per 16-row Q tile ----------------
// Q: [b*h][LQ][64], K: [b*h][LK][64], Vt: [b*h][64][LK] (transposed), ctx: [b][LQ][512]
__global__ void __launch_bounds__(128)
k_attn(const _Float16* __restrict__ Q, const _Float16* __restrict__ K,
       const _Float16* __restrict__ Vt, _Float16* __restrict__ ctx) {
  __shared__ _Float16 ldsP[4][16 * 32];   // per-wave C->A relayout staging
  const int lane = threadIdx.x & 31, wib = threadIdx.x >> 5;
  const int w = blockIdx.x * 4 + wib;
  const int qtile = w & 127, bh = w >> 7;
  const int b = bh >> 3, h = bh & 7;
  const int lane16 = lane & 15, grp = lane >> 4;

  // Q tile as two A fragments (dk = 64 = 2 x 32 contraction chunks)
  AFrag qa[2];
  const _Float16* qp = Q + ((size_t)bh * LQ + qtile * 16 + lane16) * DK;
#pragma unroll
  for (int c = 0; c < 2; ++c) {
    qa[c].h[0] = *(const v8h*)(qp + c * 32 + 16 * grp);
    qa[c].h[1] = *(const v8h*)(qp + c * 32 + 16 * grp + 8);
  }

  v8f z = {};
  v8f o[4];
#pragma unroll
  for (int t = 0; t < 4; ++t) o[t] = z;
  float mrow[8], lrow[8];
#pragma unroll
  for (int i = 0; i < 8; ++i) { mrow[i] = -1e30f; lrow[i] = 0.0f; }

  const _Float16* kbase = K + (size_t)bh * LK * DK;
  const _Float16* vbase = Vt + (size_t)bh * DK * LK;
  _Float16* lp = ldsP[wib];

  for (int kt = 0; kt < LK / 32; ++kt) {
    const _Float16* kp = kbase + (size_t)kt * 32 * DK;
    const _Float16* vp = vbase + kt * 32 + 16 * grp;
    __builtin_prefetch(kp + 32 * DK, 0, 1);   // global_prefetch next K chunk

    // Issue ALL global loads for this chunk up front (K and V fragments) so
    // their latency overlaps the S-WMMAs and the softmax VALU/LDS work.
    v16h kb[2][2];  // [contraction chunk][key tile]
#pragma unroll
    for (int t = 0; t < 2; ++t)
#pragma unroll
      for (int c = 0; c < 2; ++c)
        kb[c][t] = *(const v16h*)(kp + (size_t)(t * 16 + lane16) * DK + c * 32 + 16 * grp);
    v16h vb[4];
#pragma unroll
    for (int t = 0; t < 4; ++t)
      vb[t] = *(const v16h*)(vp + (size_t)(t * 16 + lane16) * LK);

    v8f s0 = z, s1 = z;
    s0 = wmma_f16(qa[0].v, kb[0][0], s0); s0 = wmma_f16(qa[1].v, kb[1][0], s0);
    s1 = wmma_f16(qa[0].v, kb[0][1], s1); s1 = wmma_f16(qa[1].v, kb[1][1], s1);

    float lm[8], nm[8], al[8], p0[8], p1[8], rs[8];
#pragma unroll
    for (int i = 0; i < 8; ++i) {
      s0[i] *= 0.125f; s1[i] *= 0.125f;   // 1/sqrt(dk)
      lm[i] = fmaxf(s0[i], s1[i]);
    }
#pragma unroll
    for (int off = 1; off < 16; off <<= 1)   // masks < 16 stay within the row-half
#pragma unroll
      for (int i = 0; i < 8; ++i)
        lm[i] = fmaxf(lm[i], __shfl_xor(lm[i], off, 32));
#pragma unroll
    for (int i = 0; i < 8; ++i) {
      nm[i] = fmaxf(mrow[i], lm[i]);
      al[i] = __expf(mrow[i] - nm[i]);
      p0[i] = __expf(s0[i] - nm[i]);
      p1[i] = __expf(s1[i] - nm[i]);
      rs[i] = p0[i] + p1[i];
      mrow[i] = nm[i];
    }
#pragma unroll
    for (int off = 1; off < 16; off <<= 1)
#pragma unroll
      for (int i = 0; i < 8; ++i)
        rs[i] += __shfl_xor(rs[i], off, 32);
#pragma unroll
    for (int i = 0; i < 8; ++i) lrow[i] = lrow[i] * al[i] + rs[i];
#pragma unroll
    for (int t = 0; t < 4; ++t)
#pragma unroll
      for (int i = 0; i < 8; ++i) o[t][i] *= al[i];

    // P: C layout -> LDS -> A layout (same-wave DS ops are in-order)
#pragma unroll
    for (int i = 0; i < 8; ++i) {
      lp[(i + 8 * grp) * 32 + lane16]      = (_Float16)p0[i];
      lp[(i + 8 * grp) * 32 + 16 + lane16] = (_Float16)p1[i];
    }
    asm volatile("s_wait_dscnt 0x0" ::: "memory");
    AFrag pa;
    const _Float16* pr = lp + lane16 * 32;
    pa.h[0] = *(const v8h*)(pr + 16 * grp);
    pa.h[1] = *(const v8h*)(pr + 16 * grp + 8);

#pragma unroll
    for (int t = 0; t < 4; ++t)
      o[t] = wmma_f16(pa.v, vb[t], o[t]);
  }

  float inv[8];
#pragma unroll
  for (int i = 0; i < 8; ++i) inv[i] = 1.0f / lrow[i];
#pragma unroll
  for (int t = 0; t < 4; ++t)
#pragma unroll
    for (int i = 0; i < 8; ++i) {
      int row = qtile * 16 + i + 8 * grp;
      int col = h * DK + t * 16 + lane16;
      ctx[((size_t)b * LQ + row) * D_ + col] = (_Float16)(o[t][i] * inv[i]);
    }
}

// ---------------- FC + LayerNorm fused: block computes 16 rows x 512 cols ----------------
__global__ void __launch_bounds__(256)
k_fc_ln(const _Float16* __restrict__ A, const _Float16* __restrict__ W,
        const float* __restrict__ bias, const float* __restrict__ g,
        const float* __restrict__ be, float* __restrict__ out) {
  __shared__ float sO[16 * D_];   // 32 KB
  const int lane = threadIdx.x & 31, wib = threadIdx.x >> 5;
  const int lane16 = lane & 15, grp = lane >> 4;
  const int mtile = blockIdx.x;

  v8f z = {};
  v8f o[4];
#pragma unroll
  for (int t = 0; t < 4; ++t) o[t] = z;

  const _Float16* ap0 = A + (size_t)(mtile * 16 + lane16) * D_;
  for (int c = 0; c < D_ / 32; ++c) {
    AFrag a;
    const _Float16* ap = ap0 + c * 32;
    a.h[0] = *(const v8h*)(ap + 16 * grp);
    a.h[1] = *(const v8h*)(ap + 16 * grp + 8);
#pragma unroll
    for (int t = 0; t < 4; ++t) {
      const _Float16* wp = W + (size_t)(wib * 64 + t * 16 + lane16) * D_ + c * 32 + 16 * grp;
      v16h bf = *(const v16h*)wp;
      o[t] = wmma_f16(a.v, bf, o[t]);
    }
  }
#pragma unroll
  for (int t = 0; t < 4; ++t) {
    int col = wib * 64 + t * 16 + lane16;
    float bv = bias[col];
#pragma unroll
    for (int i = 0; i < 8; ++i)
      sO[(i + 8 * grp) * D_ + col] = o[t][i] + bv;
  }
  __syncthreads();

  // LayerNorm: 16 threads per row (a half-wave), 32 cols each
  const int r = threadIdx.x >> 4, seg = threadIdx.x & 15;
  float sum = 0.f, sq = 0.f;
#pragma unroll
  for (int j = 0; j < 32; ++j) {
    float x = sO[r * D_ + seg * 32 + j];
    sum += x; sq += x * x;
  }
#pragma unroll
  for (int off = 1; off < 16; off <<= 1) {
    sum += __shfl_xor(sum, off, 32);
    sq  += __shfl_xor(sq,  off, 32);
  }
  float mu = sum * (1.0f / 512.0f);
  float var = sq * (1.0f / 512.0f) - mu * mu;
  float rstd = rsqrtf(var + 1e-5f);
  float* op = out + ((size_t)mtile * 16 + r) * D_;
#pragma unroll
  for (int j = 0; j < 32; ++j) {
    int col = seg * 32 + j;
    float x = sO[r * D_ + col];
    op[col] = (x - mu) * rstd * g[col] + be[col];
  }
}

extern "C" void kernel_launch(void* const* d_in, const int* in_sizes, int n_in,
                              void* d_out, int out_size, void* d_ws, size_t ws_size,
                              hipStream_t stream) {
  const float* enc  = (const float*)d_in[0];
  const int*   pos  = (const int*)d_in[1];
  const float* w_q  = (const float*)d_in[2];
  const float* b_q  = (const float*)d_in[3];
  const float* w_k  = (const float*)d_in[4];
  const float* b_k  = (const float*)d_in[5];
  const float* w_v  = (const float*)d_in[6];
  const float* b_v  = (const float*)d_in[7];
  const float* w_fc = (const float*)d_in[8];
  const float* b_fc = (const float*)d_in[9];
  const float* ln_g = (const float*)d_in[10];
  const float* ln_b = (const float*)d_in[11];
  const float* ptab = (const float*)d_in[12];
  (void)in_sizes; (void)n_in; (void)out_size; (void)ws_size;

  char* ws = (char*)d_ws;
  size_t off = 0;
  auto alloc16 = [&](size_t nElem) {
    _Float16* p = (_Float16*)(ws + off);
    off += nElem * sizeof(_Float16);
    return p;
  };
  _Float16* encH = alloc16((size_t)B_ * LK * D_);
  _Float16* qinH = alloc16((size_t)B_ * LQ * D_);
  _Float16* wqH  = alloc16((size_t)D_ * D_);
  _Float16* wkH  = alloc16((size_t)D_ * D_);
  _Float16* wvH  = alloc16((size_t)D_ * D_);
  _Float16* wfcH = alloc16((size_t)D_ * D_);
  _Float16* Qh   = alloc16((size_t)B_ * H_ * LQ * DK);
  _Float16* Kh   = alloc16((size_t)B_ * H_ * LK * DK);
  _Float16* Vth  = alloc16((size_t)B_ * H_ * DK * LK);
  _Float16* ctxH = alloc16((size_t)B_ * LQ * D_);

  k_cvt_f32_f16<<<(B_ * LK * D_) / 256, 256, 0, stream>>>(enc, encH, B_ * LK * D_);
  k_cvt_f32_f16<<<(D_ * D_) / 256, 256, 0, stream>>>(w_q, wqH, D_ * D_);
  k_cvt_f32_f16<<<(D_ * D_) / 256, 256, 0, stream>>>(w_k, wkH, D_ * D_);
  k_cvt_f32_f16<<<(D_ * D_) / 256, 256, 0, stream>>>(w_v, wvH, D_ * D_);
  k_cvt_f32_f16<<<(D_ * D_) / 256, 256, 0, stream>>>(w_fc, wfcH, D_ * D_);
  k_gather_pos<<<(B_ * LQ * D_) / 256, 256, 0, stream>>>(ptab, pos, qinH);

  // Q/K row-major per-head; V transposed per-head for contiguous B-fragments
  k_proj16<<<((B_ * LQ / 16) * 8) / 4, 128, 0, stream>>>(qinH, wqH, b_q, Qh, 11, LQ * DK, DK, 1);
  k_proj16<<<((B_ * LK / 16) * 8) / 4, 128, 0, stream>>>(encH, wkH, b_k, Kh, 10, LK * DK, DK, 1);
  k_proj16<<<((B_ * LK / 16) * 8) / 4, 128, 0, stream>>>(encH, wvH, b_v, Vth, 10, DK * LK, 1, LK);

  k_attn<<<(B_ * H_ * (LQ / 16)) / 4, 128, 0, stream>>>(Qh, Kh, Vth, ctxH);
  k_fc_ln<<<(B_ * LQ) / 16, 256, 0, stream>>>(ctxH, wfcH, b_fc, ln_g, ln_b, (float*)d_out);
}
